// GPTQLinear_woo_15479062135381
// MI455X (gfx1250) — matrix operand order repaired and checked by reading, not compiled, not run on hardware
//
#include <hip/hip_runtime.h>
#include <hip/hip_bf16.h>

typedef __attribute__((ext_vector_type(16))) _Float16     v16h;
typedef __attribute__((ext_vector_type(2)))  _Float16     hh2;
typedef __attribute__((ext_vector_type(8)))  float        v8f;
typedef __attribute__((ext_vector_type(8)))  unsigned int v8u;

#define IN_F    4096
#define OUT_F   11008
#define BATCH   128
#define NGROUPS 32      // IN_F / 128 (GROUP_SIZE)
#define GRP_BLK 16      // groups per block (2-way K split)
#define QROWS   512     // IN_F/32*4
#define NT_WG   64      // output columns per workgroup
#define KCHUNK  128     // K per stage == one quant group

// Grid (172, 2): blockIdx.x -> 64-column strip, blockIdx.y -> K half.
// One workgroup = 8 wave32. Wave w computes rows [16w,16w+16) x 64 cols.
// Stage = one quant group (K=128): 256 threads dequantize the 128x64 B tile
// into LDS in WMMA-B fragment layout (nibble->f16 magic + v_pk_fma_f16),
// A fragments for all 4 ksteps are loaded/converted before the barrier,
// then each wave runs 4 ksteps x 4 ntiles of v_wmma_f32_16x16x32_f16.
// Epilogue: global_atomic_add_f32 (2 deterministic addends per element).
__global__ __launch_bounds__(256)
void gptq_wmma_f16_kernel(const float* __restrict__ x,
                          const int*   __restrict__ qw,
                          const float* __restrict__ scales,
                          const float* __restrict__ zeros,
                          float* __restrict__ out)
{
    // 2 buffers x [4 ksteps x 4 ntiles x 32 lanes] x 32B fragment chunks = 32KB
    __shared__ v8u ldsB[1024];

    const int tid   = threadIdx.x;
    const int lane  = tid & 31;
    const int wv    = tid >> 5;            // wave id 0..7 -> M tile
    const int nbase = blockIdx.x * NT_WG;  // 172 blocks * 64 = 11008
    const int g0    = blockIdx.y * GRP_BLK;
    const int gend  = g0 + GRP_BLK;
    const int rend  = gend * 16;           // last qweight row + 1 for this block

    // dequant role: 4 threads per column, 4 qweight words (rows) each
    const int dq_col  = tid & 63;          // local column 0..63
    const int dq_widx = tid >> 6;          // 0..3 == kstep it fills
    const int colg    = nbase + dq_col;    // global column

    v8f acc[4];
#pragma unroll
    for (int nt = 0; nt < 4; ++nt)
        acc[nt] = (v8f){0.f, 0.f, 0.f, 0.f, 0.f, 0.f, 0.f, 0.f};

    // A-side addressing: lane l(0..15) holds row 16w+l, K offsets +0..7,+16..23;
    // lane l+16 same row, K offsets +8..15,+24..31 (16-bit A layout).
    const float* xrow = x + (size_t)(wv * 16 + (lane & 15)) * IN_F;
    const int    klo  = (lane >> 4) * 8;

    for (int g = g0; g < gend; ++g) {
        const int buf = (g & 1) * 512;

        // ---------------- dequantize B tile for this group into LDS ----------
        {
            const float s  = scales[(size_t)g * OUT_F + colg];
            const float z  = zeros [(size_t)g * OUT_F + colg];
            // nibble trick: f16(0x6400|q) == 1024+q  =>  w = s*f - (1024*s + z)
            const _Float16 sh  = (_Float16)s;
            const _Float16 nzh = (_Float16)(-fmaf(1024.0f, s, z));
            const hh2 s2  = (hh2){sh, sh};
            const hh2 nz2 = (hh2){nzh, nzh};

#pragma unroll
            for (int i = 0; i < 4; ++i) {
                const int rr = dq_widx * 4 + i;      // row within chunk 0..15
                const int r  = g * 16 + rr;          // global qweight row
                const unsigned int q =
                    (unsigned int)qw[(size_t)r * OUT_F + colg];
                if (r + 16 < rend)                   // prefetch next stage
                    __builtin_prefetch(&qw[(size_t)(r + 16) * OUT_F + colg], 0, 1);

                unsigned int pk[4];
#pragma unroll
                for (int j = 0; j < 4; ++j) {
                    const unsigned int lo = (q >> (8 * j))     & 0xFu;
                    const unsigned int hi = (q >> (8 * j + 4)) & 0xFu;
                    const unsigned int p  = 0x64006400u | lo | (hi << 16);
                    const hh2 ph = __builtin_bit_cast(hh2, p);
                    const hh2 w  = ph * s2 + nz2;    // v_pk_fma_f16
                    pk[j] = __builtin_bit_cast(unsigned int, w);
                }
                // B fragment layout: lane = n(0..15) [+16 if K%16>=8],
                // halves 0..7 = K%32<16, 8..15 = K%32>=16; this word covers 8 consecutive K -> one b128.
                const int fl  = (dq_col & 15) + ((rr & 1) ? 16 : 0);
                const int nt  = dq_col >> 4;
                const int idx = buf + (dq_widx * 4 + nt) * 32 + fl;
                uint4* dst = (uint4*)&ldsB[idx];
                dst[(rr >> 1) & 1] = make_uint4(pk[0], pk[1], pk[2], pk[3]);
            }
        }

        // --------- A fragments for all 4 ksteps (pre-barrier: overlaps) ------
        const int k0 = g * KCHUNK;
        v16h afrag[4];
#pragma unroll
        for (int ks = 0; ks < 4; ++ks) {
            const float* xp = xrow + k0 + ks * 32 + klo;
            const float4 f0 = *(const float4*)(xp);
            const float4 f1 = *(const float4*)(xp + 4);
            const float4 f2 = *(const float4*)(xp + 16);
            const float4 f3 = *(const float4*)(xp + 20);
            v16h a;
            a[0]  = (_Float16)f0.x; a[1]  = (_Float16)f0.y;
            a[2]  = (_Float16)f0.z; a[3]  = (_Float16)f0.w;
            a[4]  = (_Float16)f1.x; a[5]  = (_Float16)f1.y;
            a[6]  = (_Float16)f1.z; a[7]  = (_Float16)f1.w;
            a[8]  = (_Float16)f2.x; a[9]  = (_Float16)f2.y;
            a[10] = (_Float16)f2.z; a[11] = (_Float16)f2.w;
            a[12] = (_Float16)f3.x; a[13] = (_Float16)f3.y;
            a[14] = (_Float16)f3.z; a[15] = (_Float16)f3.w;
            afrag[ks] = a;
        }

        __syncthreads();   // single barrier/stage; double buffering covers WAR

        // ---------------- WMMA compute over this K chunk ----------------------
#pragma unroll
        for (int ks = 0; ks < 4; ++ks) {
#pragma unroll
            for (int nt = 0; nt < 4; ++nt) {
                const v16h b =
                    __builtin_bit_cast(v16h, ldsB[buf + (ks * 4 + nt) * 32 + lane]);
                acc[nt] = __builtin_amdgcn_wmma_f32_16x16x32_f16(
                    /*neg_a=*/false, afrag[ks], /*neg_b=*/false, b,
                    /*c_mod=*/(short)0, acc[nt],
                    /*reuse_a=*/false, /*reuse_b=*/false);
            }
        }
    }

    // ---------------- epilogue: f32 C/D layout, atomic K-split reduce --------
    // lanes 0-15: VGPR e -> M=e, N=lane; lanes 16-31: M=8+e, N=lane-16
    const int crow = wv * 16 + (lane >> 4) * 8;
    const int ccol = nbase + (lane & 15);
#pragma unroll
    for (int nt = 0; nt < 4; ++nt) {
#pragma unroll
        for (int e = 0; e < 8; ++e) {
            float* p = out + (size_t)(crow + e) * OUT_F + ccol + nt * 16;
            (void)__hip_atomic_fetch_add(p, acc[nt][e],
                                         __ATOMIC_RELAXED,
                                         __HIP_MEMORY_SCOPE_AGENT);
        }
    }
}

extern "C" void kernel_launch(void* const* d_in, const int* in_sizes, int n_in,
                              void* d_out, int out_size, void* d_ws, size_t ws_size,
                              hipStream_t stream) {
    (void)in_sizes; (void)n_in; (void)d_ws; (void)ws_size;
    const float* x      = (const float*)d_in[0];
    const int*   qw     = (const int*)  d_in[1];
    const float* scales = (const float*)d_in[2];
    const float* zeros  = (const float*)d_in[3];
    float*       out    = (float*)d_out;

    // K-split accumulates into out: zero it every call (graph-capturable).
    hipMemsetAsync(out, 0, (size_t)out_size * sizeof(float), stream);

    dim3 grid(OUT_F / NT_WG, NGROUPS / GRP_BLK);  // (172, 2)
    dim3 block(256);                              // 8 wave32
    gptq_wmma_f16_kernel<<<grid, block, 0, stream>>>(x, qw, scales, zeros, out);
}
